// GRUAE_89215060672656
// MI455X (gfx1250) — matrix operand order repaired and checked by reading, not compiled, not run on hardware
//
#include <hip/hip_runtime.h>
#include <hip/hip_bf16.h>

#define SEQ 4096

typedef __attribute__((ext_vector_type(16))) __bf16 v16bf;
typedef __attribute__((ext_vector_type(8)))  float  v8f;

union ABu {
  v16bf v;
  unsigned u[8];
  uint4 q[2];
};

__device__ __forceinline__ unsigned short f2bf(float f) {
  unsigned u = __float_as_uint(f);
  u += 0x7fffu + ((u >> 16) & 1u);
  return (unsigned short)(u >> 16);
}

__device__ __forceinline__ v8f wmma_bf16(ABu a, ABu b, v8f c) {
  // D(f32 16x16) = A(bf16 16x32) * B(bf16 32x16) + C
  return __builtin_amdgcn_wmma_f32_16x16x32_bf16(false, a.v, false, b.v,
                                                 (short)0, c, false, false);
}

// ---------------------------------------------------------------------------
// Zero the grid-barrier counters (re-run on every call / graph replay).
__global__ void init_counters_kernel(unsigned* cnt) {
  if (threadIdx.x < 16) cnt[threadIdx.x] = 0u;
}

// ---------------------------------------------------------------------------
// Pack fp32 row-major W[Nrows, K] into the WMMA B-operand register layout:
// out[((ntile*(K/32)+ktile)*32 + lane)*16 + i] =
//   bf16( W[ntile*16 + (lane&15), ktile*32 + (lane>=16 ? 16 : 0) + i] )
// (lane = output column N, 16 contiguous K-halves per lane)
__global__ void pack_w_kernel(const float* __restrict__ W,
                              unsigned short* __restrict__ out,
                              int Nrows, int K) {
  long gid = (long)blockIdx.x * blockDim.x + threadIdx.x;
  long ngroups = (long)(Nrows / 16) * (K / 32) * 32;
  if (gid >= ngroups) return;
  int lane = (int)(gid & 31);
  long tile = gid >> 5;
  int kt = (int)(tile % (K / 32));
  int nt = (int)(tile / (K / 32));
  int r = nt * 16 + (lane & 15);
  int kbase = kt * 32 + ((lane & 16) ? 16 : 0);
  const float* src = W + (size_t)r * K + kbase;
  unsigned short* dst = out + (size_t)gid * 16;
#pragma unroll
  for (int i = 0; i < 16; ++i) dst[i] = f2bf(src[i]);
}

// ---------------------------------------------------------------------------
// WMMA GEMM: C[T, N] = Xbf[T, K] * Wpacked^T + bias  (fp32 accum/out)
// grid = (N/64, T/128), block = 256 (8 waves).
// Each wave computes a 16(M) x 64(N) strip: 4 accumulators share one A fetch,
// cutting X L2 traffic 4x and raising WMMA density in the inner loop.
__global__ void wmma_gemm_xg_kernel(const unsigned short* __restrict__ Xbf,
                                    const unsigned short* __restrict__ Wp,
                                    const float* __restrict__ bias,
                                    float* __restrict__ C,
                                    int K, int N) {
  const int lane = threadIdx.x & 31;
  const int wave = threadIdx.x >> 5;
  const int n0 = blockIdx.x * 64;
  const int m0 = blockIdx.y * 128 + wave * 16;
  const int Kt = K >> 5;

  const int row = m0 + (lane & 15);
  const unsigned* Xu = (const unsigned*)(Xbf + (size_t)row * K);
  const int o = (lane & 16) ? 8 : 0; // A-operand dword sub-pattern select
  const size_t ntStride = (size_t)Kt * 256; // dwords between packed n-tiles
  const unsigned* Bu =
      (const unsigned*)Wp + (size_t)(blockIdx.x * 4) * ntStride + lane * 8;

  v8f c0 = {0.f, 0.f, 0.f, 0.f, 0.f, 0.f, 0.f, 0.f};
  v8f c1 = c0, c2 = c0, c3 = c0;
  for (int kt = 0; kt < Kt; ++kt) {
    ABu a, b0, b1, b2, b3;
    const int kb = kt * 16;
    a.q[0] = *(const uint4*)(Xu + kb + o);
    a.q[1] = *(const uint4*)(Xu + kb + o + 4);
    const unsigned* bp = Bu + (size_t)kt * 256;
    b0.q[0] = *(const uint4*)(bp);
    b0.q[1] = *(const uint4*)(bp + 4);
    b1.q[0] = *(const uint4*)(bp + ntStride);
    b1.q[1] = *(const uint4*)(bp + ntStride + 4);
    b2.q[0] = *(const uint4*)(bp + 2 * ntStride);
    b2.q[1] = *(const uint4*)(bp + 2 * ntStride + 4);
    b3.q[0] = *(const uint4*)(bp + 3 * ntStride);
    b3.q[1] = *(const uint4*)(bp + 3 * ntStride + 4);
    c0 = wmma_bf16(a, b0, c0);
    c1 = wmma_bf16(a, b1, c1);
    c2 = wmma_bf16(a, b2, c2);
    c3 = wmma_bf16(a, b3, c3);
  }
  const int nc = lane & 15;
  const int mbase = m0 + ((lane & 16) ? 8 : 0);
#pragma unroll
  for (int t4 = 0; t4 < 4; ++t4) {
    const v8f* cc = (t4 == 0) ? &c0 : (t4 == 1) ? &c1 : (t4 == 2) ? &c2 : &c3;
    const int ncol = n0 + t4 * 16 + nc;
    const float bv = bias[ncol];
#pragma unroll
    for (int i = 0; i < 8; ++i)
      C[(size_t)(mbase + i) * N + ncol] = (*cc)[i] + bv;
  }
}

// ---------------------------------------------------------------------------
// Persistent GRU scan. One wave owns 16 hidden units (j-tile); per step each
// wave runs 3 WMMA K-accumulation chains (r/z/n gates) against L2-resident
// packed Whh, with h replicated across the 16 A-rows from an LDS bf16 copy.
// Device-scope barrier between timesteps.
#define MODE_E1 0
#define MODE_XG 1

__global__ void gru_scan_kernel(int H, int mode,
                                const float* __restrict__ x,       // MODE_E1: [SEQ]
                                const float* __restrict__ wih_vec, // MODE_E1: [3H]
                                const float* __restrict__ bih,     // MODE_E1: [3H]
                                const float* __restrict__ xg,      // MODE_XG
                                long xg_stride,
                                const unsigned short* __restrict__ whh_packed,
                                const float* __restrict__ bhh,     // [3H]
                                float* __restrict__ hbuf,          // [2][H]
                                unsigned short* __restrict__ ys_bf,// [SEQ,H] or null
                                float* __restrict__ ys_f32,        // [SEQ,H] or null
                                float* __restrict__ emb_out,       // [H] or null
                                unsigned* __restrict__ barrier_cnt,
                                int nblocks) {
  __shared__ unsigned short sh_h[2048]; // bf16 copy of h (H <= 2048)
  const int tid = threadIdx.x;
  const int lane = tid & 31;
  const int wave = tid >> 5;
  const int jt = blockIdx.x * 8 + wave; // 16-row hidden tile owned by wave
  const int Kt = H >> 5;                // K tiles of 32
  const int Ht16 = H >> 4;              // row tiles per gate
  const int o = (lane & 16) ? 8 : 0;

  const unsigned* pw = (const unsigned*)whh_packed;
  const unsigned* bR = pw + (((size_t)(0 * Ht16 + jt) * Kt) * 32 + lane) * 8;
  const unsigned* bZ = pw + (((size_t)(1 * Ht16 + jt) * Kt) * 32 + lane) * 8;
  const unsigned* bN = pw + (((size_t)(2 * Ht16 + jt) * Kt) * 32 + lane) * 8;

  for (int t = 0; t < SEQ; ++t) {
    // Stage h (bf16) into LDS
    if (t == 0) {
      for (int i = tid; i < (H >> 1); i += blockDim.x)
        ((unsigned*)sh_h)[i] = 0u;
    } else {
      const float* hc = hbuf + (size_t)(t & 1) * H;
      for (int i = tid; i < H; i += blockDim.x) sh_h[i] = f2bf(hc[i]);
    }
    __syncthreads();

    const unsigned* shu = (const unsigned*)sh_h;
    v8f cr = {0.f, 0.f, 0.f, 0.f, 0.f, 0.f, 0.f, 0.f};
    v8f cz = cr, cn = cr;
    for (int kt = 0; kt < Kt; ++kt) {
      ABu a, b1, b2, b3;
      const int kb = kt * 16;
      // replicated-row A operand built from LDS h
      a.q[0] = *(const uint4*)(shu + kb + o);
      a.q[1] = *(const uint4*)(shu + kb + o + 4);
      const size_t ko = (size_t)kt * 256;
      b1.q[0] = *(const uint4*)(bR + ko);
      b1.q[1] = *(const uint4*)(bR + ko + 4);
      b2.q[0] = *(const uint4*)(bZ + ko);
      b2.q[1] = *(const uint4*)(bZ + ko + 4);
      b3.q[0] = *(const uint4*)(bN + ko);
      b3.q[1] = *(const uint4*)(bN + ko + 4);
      if (kt + 1 < Kt) { // keep the L2 weight stream ahead
        __builtin_prefetch(bR + ko + 256, 0, 0);
        __builtin_prefetch(bZ + ko + 256, 0, 0);
        __builtin_prefetch(bN + ko + 256, 0, 0);
      }
      cr = wmma_bf16(a, b1, cr);
      cz = wmma_bf16(a, b2, cz);
      cn = wmma_bf16(a, b3, cn);
    }

    // Gate epilogue: all A-rows identical -> row-0 result lives in c[0]
    float hr = cr[0], hz = cz[0], hn = cn[0];
    if (lane < 16) {
      const int j = jt * 16 + lane;
      float xr, xz, xn;
      if (mode == MODE_E1) {
        const float xt = x[t];
        xr = xt * wih_vec[j] + bih[j];
        xz = xt * wih_vec[H + j] + bih[H + j];
        xn = xt * wih_vec[2 * H + j] + bih[2 * H + j];
      } else {
        const float* row = xg + (size_t)t * xg_stride;
        xr = row[j];
        xz = row[H + j];
        xn = row[2 * H + j];
      }
      hr += bhh[j];
      hz += bhh[H + j];
      hn += bhh[2 * H + j];
      const float r = 1.f / (1.f + __expf(-(xr + hr)));
      const float z = 1.f / (1.f + __expf(-(xz + hz)));
      const float n = tanhf(xn + r * hn);
      const float hold = (t == 0) ? 0.f : hbuf[(size_t)(t & 1) * H + j];
      const float hnew = (1.f - z) * n + z * hold;
      hbuf[(size_t)((t + 1) & 1) * H + j] = hnew;
      if (ys_bf) ys_bf[(size_t)t * H + j] = f2bf(hnew);
      if (ys_f32) ys_f32[(size_t)t * H + j] = hnew;
      if (emb_out && t == SEQ - 1) emb_out[j] = hnew;
    }

    // Device-scope barrier between timesteps (monotonic counter)
    __threadfence();
    __syncthreads();
    if (tid == 0) {
      __hip_atomic_fetch_add(barrier_cnt, 1u, __ATOMIC_RELEASE,
                             __HIP_MEMORY_SCOPE_AGENT);
      const unsigned target = (unsigned)nblocks * (unsigned)(t + 1);
      while (__hip_atomic_load(barrier_cnt, __ATOMIC_ACQUIRE,
                               __HIP_MEMORY_SCOPE_AGENT) < target)
        __builtin_amdgcn_s_sleep(2);
    }
    __syncthreads();
    __threadfence();
  }
}

// ---------------------------------------------------------------------------
// Wave-per-row matvec: out[r] = dot(W[r,:K], v) + bias[r*bias_stride]
__global__ void matvec_kernel(const float* __restrict__ W,
                              const float* __restrict__ v,
                              const float* __restrict__ bias, long bias_stride,
                              float* __restrict__ out, int Nrows, int K) {
  const int row = blockIdx.x * 8 + (threadIdx.x >> 5);
  if (row >= Nrows) return;
  const int lane = threadIdx.x & 31;
  const float* wr = W + (size_t)row * K;
  float acc = 0.f;
  for (int k = lane; k < K; k += 32) acc += wr[k] * v[k];
#pragma unroll
  for (int off = 16; off > 0; off >>= 1) acc += __shfl_down(acc, off, 32);
  if (lane == 0) out[row] = acc + bias[(size_t)row * bias_stride];
}

// ---------------------------------------------------------------------------
extern "C" void kernel_launch(void* const* d_in, const int* in_sizes, int n_in,
                              void* d_out, int out_size, void* d_ws,
                              size_t ws_size, hipStream_t stream) {
  (void)in_sizes; (void)n_in; (void)out_size; (void)ws_size;
  const float* x       = (const float*)d_in[0];
  const float* e1_Wih  = (const float*)d_in[1];
  const float* e1_Whh  = (const float*)d_in[2];
  const float* e1_bih  = (const float*)d_in[3];
  const float* e1_bhh  = (const float*)d_in[4];
  const float* e2_Wih  = (const float*)d_in[5];
  const float* e2_Whh  = (const float*)d_in[6];
  const float* e2_bih  = (const float*)d_in[7];
  const float* e2_bhh  = (const float*)d_in[8];
  const float* d1_Wih  = (const float*)d_in[9];
  const float* d1_Whh  = (const float*)d_in[10];
  const float* d1_bih  = (const float*)d_in[11];
  const float* d1_bhh  = (const float*)d_in[12];
  const float* d2_Wih  = (const float*)d_in[13];
  const float* d2_Whh  = (const float*)d_in[14];
  const float* d2_bih  = (const float*)d_in[15];
  const float* d2_bhh  = (const float*)d_in[16];
  const float* out_W   = (const float*)d_in[17];
  const float* out_b   = (const float*)d_in[18];
  float* out = (float*)d_out;

  char* ws = (char*)d_ws;
  size_t off = 0;
  auto alloc = [&](size_t bytes) -> void* {
    off = (off + 255) & ~(size_t)255;
    void* p = ws + off;
    off += bytes;
    return p;
  };

  unsigned* cnt       = (unsigned*)alloc(256);
  float* hbuf         = (float*)alloc(2 * 2048 * sizeof(float));
  float* emb          = (float*)alloc(1024 * sizeof(float));
  float* xg_d1        = (float*)alloc(3072 * sizeof(float));
  unsigned short* pWe1hh = (unsigned short*)alloc((size_t)6144 * 2048 * 2);
  unsigned short* pWe2ih = (unsigned short*)alloc((size_t)3072 * 2048 * 2);
  unsigned short* pWe2hh = (unsigned short*)alloc((size_t)3072 * 1024 * 2);
  unsigned short* pWd1hh = (unsigned short*)alloc((size_t)3072 * 1024 * 2);
  unsigned short* pWd2ih = (unsigned short*)alloc((size_t)6144 * 1024 * 2);
  unsigned short* pWd2hh = (unsigned short*)alloc((size_t)6144 * 2048 * 2);
  unsigned short* ys1_bf = (unsigned short*)alloc((size_t)SEQ * 2048 * 2);
  unsigned short* ys2_bf = (unsigned short*)alloc((size_t)SEQ * 1024 * 2);
  float* xg_e2        = (float*)alloc((size_t)SEQ * 3072 * sizeof(float));
  float* xg_d2        = (float*)alloc((size_t)SEQ * 6144 * sizeof(float));
  float* ys3          = (float*)alloc((size_t)SEQ * 2048 * sizeof(float));

  dim3 blk(256);

  init_counters_kernel<<<1, 64, 0, stream>>>(cnt);

  auto pack = [&](const float* W, unsigned short* dst, int Nr, int K) {
    long groups = (long)Nr * K / 16;
    pack_w_kernel<<<dim3((unsigned)((groups + 255) / 256)), blk, 0, stream>>>(
        W, dst, Nr, K);
  };
  pack(e1_Whh, pWe1hh, 6144, 2048);
  pack(e2_Wih, pWe2ih, 3072, 2048);
  pack(e2_Whh, pWe2hh, 3072, 1024);
  pack(d1_Whh, pWd1hh, 3072, 1024);
  pack(d2_Wih, pWd2ih, 6144, 1024);
  pack(d2_Whh, pWd2hh, 6144, 2048);

  // e1 scan: H=2048, xg on the fly from scalar input
  gru_scan_kernel<<<16, blk, 0, stream>>>(
      2048, MODE_E1, x, e1_Wih, e1_bih, nullptr, 0, pWe1hh, e1_bhh, hbuf,
      ys1_bf, nullptr, nullptr, cnt + 0, 16);

  // e2 xg GEMM: [4096,3072] = ys1_bf[4096,2048] * Wih^T + bih
  wmma_gemm_xg_kernel<<<dim3(3072 / 64, SEQ / 128), blk, 0, stream>>>(
      ys1_bf, pWe2ih, e2_bih, xg_e2, 2048, 3072);

  // e2 scan: H=1024, produce embedding h_T
  gru_scan_kernel<<<8, blk, 0, stream>>>(
      1024, MODE_XG, nullptr, nullptr, nullptr, xg_e2, 3072, pWe2hh, e2_bhh,
      hbuf, nullptr, nullptr, emb, cnt + 1, 8);

  // d1 constant xg: [3072] = d1_Wih * emb + d1_bih
  matvec_kernel<<<3072 / 8, blk, 0, stream>>>(d1_Wih, emb, d1_bih, 1, xg_d1,
                                              3072, 1024);

  // d1 scan: H=1024, xg stride 0 (constant input)
  gru_scan_kernel<<<8, blk, 0, stream>>>(
      1024, MODE_XG, nullptr, nullptr, nullptr, xg_d1, 0, pWd1hh, d1_bhh, hbuf,
      ys2_bf, nullptr, nullptr, cnt + 2, 8);

  // d2 xg GEMM: [4096,6144] = ys2_bf[4096,1024] * Wih^T + bih
  wmma_gemm_xg_kernel<<<dim3(6144 / 64, SEQ / 128), blk, 0, stream>>>(
      ys2_bf, pWd2ih, d2_bih, xg_d2, 1024, 6144);

  // d2 scan: H=2048, write ys3 fp32
  gru_scan_kernel<<<16, blk, 0, stream>>>(
      2048, MODE_XG, nullptr, nullptr, nullptr, xg_d2, 6144, pWd2hh, d2_bhh,
      hbuf, nullptr, ys3, nullptr, cnt + 3, 16);

  // final: out[t] = dot(ys3[t,:], out_W) + out_b
  matvec_kernel<<<SEQ / 8, blk, 0, stream>>>(ys3, out_W, out_b, 0, out, SEQ,
                                             2048);
}
